// Lenet_618475291234
// MI455X (gfx1250) — matrix operand order repaired and checked by reading, not compiled, not run on hardware
//
#include <hip/hip_runtime.h>

typedef _Float16 v16h __attribute__((ext_vector_type(16)));
typedef _Float16 h8   __attribute__((ext_vector_type(8)));
typedef float    v8f  __attribute__((ext_vector_type(8)));
typedef float    v2f  __attribute__((ext_vector_type(2)));

#define BN_EPS 1e-5f

__device__ __forceinline__ float sgn(float v) {
    return (v > 0.f) ? 1.f : ((v < 0.f) ? -1.f : 0.f);
}

// ---------------------------------------------------------------------------
// Stage 0: fp32 conv (1->32ch, 3x3, pad 1) + 2x2 maxpool + BN stat accumulation
// One image per block, 256 threads.
// ---------------------------------------------------------------------------
__global__ __launch_bounds__(256) void k_conv0(
    const float* __restrict__ x, const float* __restrict__ w0,
    float* __restrict__ y0, float* __restrict__ g_sum, float* __restrict__ g_ssq) {
    __shared__ float xp[30 * 30];
    __shared__ float wl[32 * 9];
    __shared__ float ssum[32], ssq[32];
    const int b = blockIdx.x, tid = threadIdx.x;
    __builtin_prefetch(x + b * 784 + (tid & 31) * 32, 0, 3);   // warm this image
    if (tid < 32) { ssum[tid] = 0.f; ssq[tid] = 0.f; }
    for (int i = tid; i < 900; i += 256) {
        int iy = i / 30 - 1, ix = i % 30 - 1;
        xp[i] = (iy >= 0 && iy < 28 && ix >= 0 && ix < 28) ? x[b * 784 + iy * 28 + ix] : 0.f;
    }
    for (int i = tid; i < 288; i += 256) wl[i] = w0[i];
    __syncthreads();
    for (int idx = tid; idx < 6272; idx += 256) {   // 32ch * 14*14 pooled outputs
        int c = idx / 196, p = idx % 196, py = p / 14, px = p % 14;
        float m = -3.4e38f;
        #pragma unroll
        for (int dy = 0; dy < 2; ++dy)
        #pragma unroll
        for (int dx = 0; dx < 2; ++dx) {
            int yy = 2 * py + dy, xx = 2 * px + dx;
            float s = 0.f;
            #pragma unroll
            for (int ky = 0; ky < 3; ++ky)
            #pragma unroll
            for (int kx = 0; kx < 3; ++kx)
                s += xp[(yy + ky) * 30 + (xx + kx)] * wl[c * 9 + ky * 3 + kx];
            m = fmaxf(m, s);
        }
        y0[b * 6272 + idx] = m;
        atomicAdd(&ssum[c], m);
        atomicAdd(&ssq[c], m * m);
    }
    __syncthreads();
    if (tid < 32) { atomicAdd(&g_sum[tid], ssum[tid]); atomicAdd(&g_ssq[tid], ssq[tid]); }
}

// ---------------------------------------------------------------------------
// BN finalize: per-channel scale/bias from accumulated sum/sumsq (biased var)
// ---------------------------------------------------------------------------
__global__ void k_bnfin(const float* __restrict__ s_sum, const float* __restrict__ s_ssq,
                        const float* __restrict__ gamma, const float* __restrict__ beta,
                        float inv_n, float* __restrict__ scale, float* __restrict__ bias) {
    int c = threadIdx.x;
    if (c < 32) {
        float mean = s_sum[c] * inv_n;
        float var  = s_ssq[c] * inv_n - mean * mean;
        float sc   = gamma[c] * rsqrtf(var + BN_EPS);
        scale[c] = sc;
        bias[c]  = beta[c] - mean * sc;
    }
}

// ---------------------------------------------------------------------------
// Stage 1: BN0-apply + sign + binary 3x3 conv via WMMA f16 (K=288=9x32) +
// maxpool + shortcut maxpool + BN1 stat accumulation. One image per block,
// 256 threads = 8 wave32. 196 pixel rows -> 13 M-tiles of 16.
// A-fragment DS loads are software-pipelined one K-step ahead of the WMMAs.
// ---------------------------------------------------------------------------
__global__ __launch_bounds__(256) void k_conv1(
    const float* __restrict__ y0, const float* __restrict__ w1,
    const float* __restrict__ scale0, const float* __restrict__ bias0,
    float* __restrict__ o1, float* __restrict__ sp1,
    float* __restrict__ g_sum, float* __restrict__ g_ssq) {
    __shared__ _Float16 shalo[16 * 16 * 32];   // [yy][xx][ic], halo=1, 16KB
    __shared__ float    hbuf[196 * 32];        // h then reused as conv output, 25KB
    __shared__ _Float16 wlds[9 * 32 * 32];     // [tap][oc][ic], 18KB
    __shared__ float    ssum[32], ssq[32];
    const int b = blockIdx.x, tid = threadIdx.x;
    __builtin_prefetch(w1 + tid * 36, 0, 3);                   // 36KB of weights
    __builtin_prefetch(y0 + b * 6272 + tid * 24, 0, 3);        // this image's input
    if (tid < 32) { ssum[tid] = 0.f; ssq[tid] = 0.f; }
    { int* zi = (int*)shalo; for (int i = tid; i < 4096; i += 256) zi[i] = 0; }
    __syncthreads();
    for (int idx = tid; idx < 6272; idx += 256) {
        int c = idx / 196, p = idx % 196, y = p / 14, xc = p % 14;
        float h = scale0[c] * y0[b * 6272 + idx] + bias0[c];
        hbuf[p * 32 + c] = h;
        shalo[((y + 1) * 16 + (xc + 1)) * 32 + c] = (_Float16)sgn(h);
    }
    for (int i = tid; i < 9216; i += 256) {          // w1: [oc][ic][ky][kx]
        int oc = i / 288, r = i % 288, ic = r / 9, tap = r % 9;
        wlds[tap * 1024 + oc * 32 + ic] = (_Float16)sgn(w1[i]);
    }
    __syncthreads();
    for (int i = tid; i < 1568; i += 256) {          // shortcut maxpool2(h) 14->7
        int c = i / 49, q = i % 49, qy = q / 7, qx = q % 7;
        float m = hbuf[((2 * qy) * 14 + 2 * qx) * 32 + c];
        m = fmaxf(m, hbuf[((2 * qy) * 14 + 2 * qx + 1) * 32 + c]);
        m = fmaxf(m, hbuf[((2 * qy + 1) * 14 + 2 * qx) * 32 + c]);
        m = fmaxf(m, hbuf[((2 * qy + 1) * 14 + 2 * qx + 1) * 32 + c]);
        sp1[b * 1568 + i] = m;
    }
    __syncthreads();                                  // hbuf now reusable as conv out
    const int wave = tid >> 5, lane = tid & 31, hlf = lane >> 4, mrow = lane & 15;
    for (int t = wave; t < 13; t += 8) {
        int p = t * 16 + mrow, pp = (p < 196) ? p : 0;
        int y = pp / 14, xc = pp % 14;
        v8f c0 = {}, c1 = {};
        h8 alo[2], ahi[2];
        {   // prologue: load K-step 0 (tap ky=0,kx=0)
            int ab = (y * 16 + xc) * 32 + hlf * 16;
            alo[0] = *(const h8*)&shalo[ab];
            ahi[0] = *(const h8*)&shalo[ab + 8];
        }
        #pragma unroll
        for (int s = 0; s < 9; ++s) {
            const int cur = s & 1, nxt = cur ^ 1;
            if (s < 8) {                              // prefetch next step's A
                int ky = (s + 1) / 3, kx = (s + 1) % 3;
                int ab = ((y + ky) * 16 + (xc + kx)) * 32 + hlf * 16;
                alo[nxt] = *(const h8*)&shalo[ab];
                ahi[nxt] = *(const h8*)&shalo[ab + 8];
            }
            v16h a;
            #pragma unroll
            for (int j = 0; j < 8; ++j) { a[j] = alo[cur][j]; a[8 + j] = ahi[cur][j]; }
            v16h b0 = *(const v16h*)&wlds[s * 1024 + mrow * 32 + hlf * 16];
            v16h b1 = *(const v16h*)&wlds[s * 1024 + (mrow + 16) * 32 + hlf * 16];
            c0 = __builtin_amdgcn_wmma_f32_16x16x32_f16(false, a, false, b0, (short)0, c0, false, false);
            c1 = __builtin_amdgcn_wmma_f32_16x16x32_f16(false, a, false, b1, (short)0, c1, false, false);
        }
        #pragma unroll
        for (int v = 0; v < 8; ++v) {                // D layout: lane n=lane%16, M=v+8*(lane/16)
            int mr = v + 8 * hlf, pr = t * 16 + mr;
            if (pr < 196) {
                hbuf[pr * 32 + mrow]      = c0[v];
                hbuf[pr * 32 + 16 + mrow] = c1[v];
            }
        }
    }
    __syncthreads();
    for (int i = tid; i < 1568; i += 256) {          // maxpool 14->7 + BN1 stats
        int c = i / 49, q = i % 49, qy = q / 7, qx = q % 7;
        float m = hbuf[((2 * qy) * 14 + 2 * qx) * 32 + c];
        m = fmaxf(m, hbuf[((2 * qy) * 14 + 2 * qx + 1) * 32 + c]);
        m = fmaxf(m, hbuf[((2 * qy + 1) * 14 + 2 * qx) * 32 + c]);
        m = fmaxf(m, hbuf[((2 * qy + 1) * 14 + 2 * qx + 1) * 32 + c]);
        o1[b * 1568 + i] = m;
        atomicAdd(&ssum[c], m);
        atomicAdd(&ssq[c], m * m);
    }
    __syncthreads();
    if (tid < 32) { atomicAdd(&g_sum[tid], ssum[tid]); atomicAdd(&g_ssq[tid], ssq[tid]); }
}

// ---------------------------------------------------------------------------
// Stage 2: BN1-apply + shortcut add + sign + binary conv (7x7) via WMMA f16 +
// floor maxpool 7->3 + shortcut + BN2 stats. One image per block, 128 threads.
// ---------------------------------------------------------------------------
__global__ __launch_bounds__(128) void k_conv2(
    const float* __restrict__ o1, const float* __restrict__ sp1,
    const float* __restrict__ w2,
    const float* __restrict__ scale1, const float* __restrict__ bias1,
    float* __restrict__ o2, float* __restrict__ sp2,
    float* __restrict__ g_sum, float* __restrict__ g_ssq) {
    __shared__ _Float16 shalo[9 * 9 * 32];
    __shared__ float    hbuf[49 * 32];
    __shared__ _Float16 wlds[9 * 32 * 32];
    __shared__ float    ssum[32], ssq[32];
    const int b = blockIdx.x, tid = threadIdx.x;
    __builtin_prefetch(w2 + tid * 72, 0, 3);
    __builtin_prefetch(o1 + b * 1568 + tid * 12, 0, 3);
    __builtin_prefetch(sp1 + b * 1568 + tid * 12, 0, 3);
    if (tid < 32) { ssum[tid] = 0.f; ssq[tid] = 0.f; }
    { int* zi = (int*)shalo; for (int i = tid; i < 1296; i += 128) zi[i] = 0; }
    __syncthreads();
    for (int idx = tid; idx < 1568; idx += 128) {
        int c = idx / 49, q = idx % 49, y = q / 7, xc = q % 7;
        float h = scale1[c] * o1[b * 1568 + idx] + bias1[c] + sp1[b * 1568 + idx];
        hbuf[q * 32 + c] = h;
        shalo[((y + 1) * 9 + (xc + 1)) * 32 + c] = (_Float16)sgn(h);
    }
    for (int i = tid; i < 9216; i += 128) {
        int oc = i / 288, r = i % 288, ic = r / 9, tap = r % 9;
        wlds[tap * 1024 + oc * 32 + ic] = (_Float16)sgn(w2[i]);
    }
    __syncthreads();
    for (int i = tid; i < 288; i += 128) {           // shortcut maxpool2 7->3 (floor)
        int c = i / 9, q = i % 9, qy = q / 3, qx = q % 3;
        float m = hbuf[((2 * qy) * 7 + 2 * qx) * 32 + c];
        m = fmaxf(m, hbuf[((2 * qy) * 7 + 2 * qx + 1) * 32 + c]);
        m = fmaxf(m, hbuf[((2 * qy + 1) * 7 + 2 * qx) * 32 + c]);
        m = fmaxf(m, hbuf[((2 * qy + 1) * 7 + 2 * qx + 1) * 32 + c]);
        sp2[b * 288 + i] = m;
    }
    __syncthreads();
    const int wave = tid >> 5, lane = tid & 31, hlf = lane >> 4, mrow = lane & 15;
    {   // 49 pixel rows -> 4 M-tiles, one per wave
        int t = wave;
        int p = t * 16 + mrow, pp = (p < 49) ? p : 0;
        int y = pp / 7, xc = pp % 7;
        v8f c0 = {}, c1 = {};
        h8 alo[2], ahi[2];
        {
            int ab = (y * 9 + xc) * 32 + hlf * 16;
            alo[0] = *(const h8*)&shalo[ab];
            ahi[0] = *(const h8*)&shalo[ab + 8];
        }
        #pragma unroll
        for (int s = 0; s < 9; ++s) {
            const int cur = s & 1, nxt = cur ^ 1;
            if (s < 8) {
                int ky = (s + 1) / 3, kx = (s + 1) % 3;
                int ab = ((y + ky) * 9 + (xc + kx)) * 32 + hlf * 16;
                alo[nxt] = *(const h8*)&shalo[ab];
                ahi[nxt] = *(const h8*)&shalo[ab + 8];
            }
            v16h a;
            #pragma unroll
            for (int j = 0; j < 8; ++j) { a[j] = alo[cur][j]; a[8 + j] = ahi[cur][j]; }
            v16h b0 = *(const v16h*)&wlds[s * 1024 + mrow * 32 + hlf * 16];
            v16h b1 = *(const v16h*)&wlds[s * 1024 + (mrow + 16) * 32 + hlf * 16];
            c0 = __builtin_amdgcn_wmma_f32_16x16x32_f16(false, a, false, b0, (short)0, c0, false, false);
            c1 = __builtin_amdgcn_wmma_f32_16x16x32_f16(false, a, false, b1, (short)0, c1, false, false);
        }
        #pragma unroll
        for (int v = 0; v < 8; ++v) {
            int mr = v + 8 * hlf, pr = t * 16 + mr;
            if (pr < 49) {
                hbuf[pr * 32 + mrow]      = c0[v];
                hbuf[pr * 32 + 16 + mrow] = c1[v];
            }
        }
    }
    __syncthreads();
    for (int i = tid; i < 288; i += 128) {           // maxpool 7->3 + BN2 stats
        int c = i / 9, q = i % 9, qy = q / 3, qx = q % 3;
        float m = hbuf[((2 * qy) * 7 + 2 * qx) * 32 + c];
        m = fmaxf(m, hbuf[((2 * qy) * 7 + 2 * qx + 1) * 32 + c]);
        m = fmaxf(m, hbuf[((2 * qy + 1) * 7 + 2 * qx) * 32 + c]);
        m = fmaxf(m, hbuf[((2 * qy + 1) * 7 + 2 * qx + 1) * 32 + c]);
        o2[b * 288 + i] = m;
        atomicAdd(&ssum[c], m);
        atomicAdd(&ssq[c], m * m);
    }
    __syncthreads();
    if (tid < 32) { atomicAdd(&g_sum[tid], ssum[tid]); atomicAdd(&g_ssq[tid], ssq[tid]); }
}

// ---------------------------------------------------------------------------
// FC head: h2 = BN2(o2)+sp2 -> [B,288]; out = h2 @ fw.T + fb via f32 WMMA
// (16x16x4, 72 K-steps). One wave per block; 16 images per block.
// ---------------------------------------------------------------------------
__global__ __launch_bounds__(32) void k_fc(
    const float* __restrict__ o2, const float* __restrict__ sp2,
    const float* __restrict__ scale2, const float* __restrict__ bias2,
    const float* __restrict__ fw, const float* __restrict__ fb,
    float* __restrict__ out) {
    __shared__ float a2[16 * 288];     // 16 images x 288 features
    __shared__ float flds[16 * 288];   // fw rows (10 real + 6 zero pad)
    const int tid = threadIdx.x, base = blockIdx.x * 16;
    __builtin_prefetch(fw + tid * 90, 0, 3);
    __builtin_prefetch(o2 + base * 288 + tid * 144, 0, 3);
    for (int i = tid; i < 16 * 288; i += 32) {
        int n = i / 288, k = i % 288;
        flds[i] = (n < 10) ? fw[n * 288 + k] : 0.f;
        int c = k / 9;   // flatten index k = c*9 + (py*3+px) matches NCHW reshape
        a2[i] = scale2[c] * o2[(base + n) * 288 + k] + bias2[c] + sp2[(base + n) * 288 + k];
    }
    __syncthreads();
    const int lane = tid & 31, hlf = lane >> 4, mrow = lane & 15;
    v8f acc = {};
    #pragma unroll 8
    for (int s = 0; s < 72; ++s) {
        int kb = 4 * s + hlf * 2;      // f32 A 16x4: lane m, K = (lane/16)*2 + v
        v2f av = { a2[mrow * 288 + kb],   a2[mrow * 288 + kb + 1] };
        v2f bv = { flds[mrow * 288 + kb], flds[mrow * 288 + kb + 1] };
        acc = __builtin_amdgcn_wmma_f32_16x16x4_f32(false, av, false, bv, (short)0, acc, false, false);
    }
    #pragma unroll
    for (int v = 0; v < 8; ++v) {
        int mr = v + 8 * hlf;
        if (mrow < 10) out[(base + mr) * 10 + mrow] = acc[v] + fb[mrow];
    }
}

extern "C" void kernel_launch(void* const* d_in, const int* in_sizes, int n_in,
                              void* d_out, int out_size, void* d_ws, size_t ws_size,
                              hipStream_t stream) {
    const float* x  = (const float*)d_in[0];
    const float* w0 = (const float*)d_in[1];
    const float* g0 = (const float*)d_in[2];
    const float* b0 = (const float*)d_in[3];
    const float* w1 = (const float*)d_in[4];
    const float* g1 = (const float*)d_in[5];
    const float* b1 = (const float*)d_in[6];
    const float* w2 = (const float*)d_in[7];
    const float* g2 = (const float*)d_in[8];
    const float* b2 = (const float*)d_in[9];
    const float* fw = (const float*)d_in[10];
    const float* fb = (const float*)d_in[11];
    float* out = (float*)d_out;
    float* ws  = (float*)d_ws;
    const int B = in_sizes[0] / 784;

    size_t off = 0;
    float* y0  = ws + off; off += (size_t)B * 6272;   // [B,32,14,14]
    float* o1  = ws + off; off += (size_t)B * 1568;   // [B,32,7,7] pre-BN
    float* sp1 = ws + off; off += (size_t)B * 1568;   // shortcut
    float* o2  = ws + off; off += (size_t)B * 288;    // [B,32,3,3] pre-BN
    float* sp2 = ws + off; off += (size_t)B * 288;
    float* st  = ws + off; off += 192;                // 3 stages x {sum[32],ssq[32]}
    float* bn  = ws + off; off += 192;                // 3 stages x {scale[32],bias[32]}

    hipMemsetAsync(st, 0, 192 * sizeof(float), stream);
    k_conv0<<<B, 256, 0, stream>>>(x, w0, y0, st + 0, st + 32);
    k_bnfin<<<1, 32, 0, stream>>>(st + 0, st + 32, g0, b0, 1.f / ((float)B * 196.f), bn + 0, bn + 32);
    k_conv1<<<B, 256, 0, stream>>>(y0, w1, bn + 0, bn + 32, o1, sp1, st + 64, st + 96);
    k_bnfin<<<1, 32, 0, stream>>>(st + 64, st + 96, g1, b1, 1.f / ((float)B * 49.f), bn + 64, bn + 96);
    k_conv2<<<B, 128, 0, stream>>>(o1, sp1, w2, bn + 64, bn + 96, o2, sp2, st + 128, st + 160);
    k_bnfin<<<1, 32, 0, stream>>>(st + 128, st + 160, g2, b2, 1.f / ((float)B * 9.f), bn + 128, bn + 160);
    k_fc<<<B / 16, 32, 0, stream>>>(o2, sp2, bn + 128, bn + 160, fw, fb, out);
}
